// MambaLayer_20255065768375
// MI455X (gfx1250) — matrix operand order repaired and checked by reading, not compiled, not run on hardware
//
#include <hip/hip_runtime.h>
#include <hip/hip_bf16.h>
#include <math.h>
#include <stdint.h>

// ---------------- problem dimensions (fixed by reference) ----------------
#define NT     8192      // BATCH*SEQ tokens
#define SEQL   2048
#define BATCHN 4
#define DMODEL 512
#define DINNER 1024
#define DSTATE 16
#define NCHUNK 32        // scan chunks per sequence
#define CLEN   64        // SEQL / NCHUNK

typedef unsigned short bf16_t;   // raw bf16 storage
typedef __attribute__((ext_vector_type(16))) __bf16 v16bf;
typedef __attribute__((ext_vector_type(8)))  float  v8f;

__device__ __forceinline__ bf16_t f2bf(float f) {
  union { float f; unsigned u; } v; v.f = f;
  return (bf16_t)((v.u + 0x7FFFu + ((v.u >> 16) & 1u)) >> 16);   // RNE
}
__device__ __forceinline__ float siluf_(float x) { return x / (1.0f + __expf(-x)); }

// CDNA5 async global->LDS copy (ASYNCcnt-tracked, no VGPR round trip).
// lds = 32-bit LDS byte offset (low 32 bits of a generic LDS pointer),
// ga  = 64-bit global address.
__device__ __forceinline__ void async_g2l_b128(unsigned lds, unsigned long long ga) {
  asm volatile("global_load_async_to_lds_b128 %0, %1, off"
               :: "v"(lds), "v"(ga) : "memory");
}
__device__ __forceinline__ void wait_async0() {
  asm volatile("s_wait_asynccnt 0" ::: "memory");
}

// =====================================================================
// WMMA GEMM:  C(MxN, f32) = A(MxK, bf16) @ B(NxK, bf16)^T  [+bias][+= C]
// Block tile 128x64, K tile 32. 128 threads = 4 wave32 in a 2x2 grid,
// each wave owns a 64x32 tile = 4x2 v_wmma_f32_16x16x32_bf16 frags.
// Double-buffered LDS fed by global_load_async_to_lds_b128: the DMA of
// tile k+1 overlaps the WMMAs of tile k; one s_wait_asynccnt + two
// barriers per K tile.  Per-thread staging descriptors are precomputed
// (branch-free, strided 64B/tile) so the inner loop has no 64-bit
// multiplies or exec-mask branches.
// Requires: M%128==0, N%64==0, K%32==0 (true for all calls here).
// flags: bit0 = add bias[col]; bit1 = accumulate into existing C.
// =====================================================================
#define BM 128
#define BN 64
#define BK 32
#define LDP 48    // LDS pitch in bf16 elems (96B, 16B-aligned rows)

union FragBF { uint4 q[2]; v16bf v; };

__global__ __launch_bounds__(128) void gemm_bf16_wmma(
    const bf16_t* __restrict__ A, const bf16_t* __restrict__ B,
    const float* __restrict__ bias, float* __restrict__ C,
    int M, int N, int K, int flags)
{
  __shared__ bf16_t As[2][BM * LDP];
  __shared__ bf16_t Bs[2][BN * LDP];
  const int tid  = threadIdx.x;
  const int lane = tid & 31;
  const int wave = tid >> 5;
  const int wm = wave >> 1, wn = wave & 1;
  const int m0 = blockIdx.y * BM, n0 = blockIdx.x * BN;
  const int hl = lane >> 4;     // half-wave select
  const int ll = lane & 15;

  // ---- precomputed per-thread staging descriptors ----
  // A tile: 128x32 bf16 = 512 16B chunks -> 4 per thread (u = 0..3)
  // B tile:  64x32 bf16 = 256 16B chunks -> 2 per thread (u = 0..1)
  unsigned           ldsA[4], ldsB[2];
  unsigned long long gaA[4],  gaB[2];
  #pragma unroll
  for (int u = 0; u < 4; ++u) {
    int i = tid + u * 128;
    int row = i >> 2, c8 = (i & 3) << 3;
    ldsA[u] = (unsigned)(uintptr_t)&As[0][row * LDP + c8];
    gaA[u]  = (unsigned long long)(uintptr_t)&A[(size_t)(m0 + row) * K + c8];
  }
  #pragma unroll
  for (int u = 0; u < 2; ++u) {
    int i = tid + u * 128;
    int row = i >> 2, c8 = (i & 3) << 3;
    ldsB[u] = (unsigned)(uintptr_t)&Bs[0][row * LDP + c8];
    gaB[u]  = (unsigned long long)(uintptr_t)&B[(size_t)(n0 + row) * K + c8];
  }
  const unsigned ASTR = BM * LDP * 2;  // byte distance As[0] -> As[1]
  const unsigned BSTR = BN * LDP * 2;  // byte distance Bs[0] -> Bs[1]

  auto stage = [&](int buf) {
    unsigned ao = buf ? ASTR : 0u, bo = buf ? BSTR : 0u;
    #pragma unroll
    for (int u = 0; u < 4; ++u) { async_g2l_b128(ldsA[u] + ao, gaA[u]); gaA[u] += BK * 2; }
    #pragma unroll
    for (int u = 0; u < 2; ++u) { async_g2l_b128(ldsB[u] + bo, gaB[u]); gaB[u] += BK * 2; }
  };

  v8f acc[4][2] = {};

  stage(0);
  const int nk = K >> 5;            // K / BK
  for (int kt = 0; kt < nk; ++kt) {
    wait_async0();                  // staged tile kt is in LDS
    __syncthreads();                // ..visible to all waves; prev reads done
    if (kt + 1 < nk) stage((kt + 1) & 1);

    const bf16_t* as = As[kt & 1];
    const bf16_t* bs = Bs[kt & 1];
    FragBF a[4], b[2];
    // A frag (16x32 bf16): lane holds row m=ll; regs0-3: k=hl*8..+7, regs4-7: +16
    #pragma unroll
    for (int sm = 0; sm < 4; ++sm) {
      int r = wm * 64 + sm * 16 + ll;
      int s = hl * 8;
      a[sm].q[0] = *(const uint4*)(&as[r * LDP + s]);
      a[sm].q[1] = *(const uint4*)(&as[r * LDP + s + 16]);
    }
    // B frag (32x16 bf16): lane holds col n=ll; regs0-3: k=hl*8..+7, regs4-7: +16
    #pragma unroll
    for (int sn = 0; sn < 2; ++sn) {
      int r = wn * 32 + sn * 16 + ll;
      int t = hl * 8;
      b[sn].q[0] = *(const uint4*)(&bs[r * LDP + t]);
      b[sn].q[1] = *(const uint4*)(&bs[r * LDP + t + 16]);
    }
    #pragma unroll
    for (int sm = 0; sm < 4; ++sm)
      #pragma unroll
      for (int sn = 0; sn < 2; ++sn)
        acc[sm][sn] = __builtin_amdgcn_wmma_f32_16x16x32_bf16(
            false, a[sm].v, false, b[sn].v, (short)0, acc[sm][sn], false, false);
    __syncthreads();
  }

  // C/D layout: reg r -> (M = r + hl*8, N = ll) within a 16x16 tile
  #pragma unroll
  for (int sm = 0; sm < 4; ++sm) {
    #pragma unroll
    for (int sn = 0; sn < 2; ++sn) {
      int col = n0 + wn * 32 + sn * 16 + ll;
      float bv = (flags & 1) ? bias[col] : 0.0f;
      int rbase = m0 + wm * 64 + sm * 16 + hl * 8;
      #pragma unroll
      for (int r = 0; r < 8; ++r) {
        size_t idx = (size_t)(rbase + r) * N + col;
        float v = acc[sm][sn][r] + bv;
        if (flags & 2) v += C[idx];
        C[idx] = v;
      }
    }
  }
}

// =====================================================================
// Circulant build: the FFT/mask/filter/IFFT(real) chain is x -> M x with
// M[i][j] = c[(i-j) mod 512],  c_m = (1/512) sum_f Re(G_f e^{2pi i f m/512})
// G_f = mask_f + re_f + i*im_f.  One block of 512 threads.
// =====================================================================
__global__ __launch_bounds__(512) void build_circ_k(
    const float* __restrict__ re, const float* __restrict__ im,
    const float* __restrict__ cutoff, bf16_t* __restrict__ Mw)
{
  __shared__ float c[DMODEL];
  int m = threadIdx.x;
  float cut = cutoff[0];
  float s = 0.0f;
  for (int f = 0; f < DMODEL; ++f) {
    float fr = (float)((f < 256) ? f : f - 512) * (1.0f / 512.0f);
    float mask = (fr >= cut || fr <= -cut) ? 1.0f : 0.0f;
    float g_re = mask + re[f];
    float g_im = im[f];
    int ph = (f * m) & 511;               // exact phase: (f*m mod 512)
    float th = 6.28318530717958647f * (float)ph * (1.0f / 512.0f);
    float sth, cth;
    __sincosf(th, &sth, &cth);
    s += g_re * cth - g_im * sth;
  }
  c[m] = s * (1.0f / 512.0f);
  __syncthreads();
  for (int j = 0; j < DMODEL; ++j)
    Mw[m * DMODEL + j] = f2bf(c[(m - j) & 511]);
}

// =====================================================================
// LayerNorm over d=512: one 256-thread block per token, 2 elems/thread.
// Optional residual input, optional bf16 mirror output.
// =====================================================================
__global__ __launch_bounds__(256) void layernorm_k(
    const float* __restrict__ in, const float* __restrict__ res,
    const float* __restrict__ g, const float* __restrict__ b,
    float* __restrict__ out, bf16_t* __restrict__ obf, float eps, int has_res)
{
  __shared__ float red[16];
  int t = blockIdx.x, tid = threadIdx.x;
  size_t base = (size_t)t * DMODEL;
  float v0 = in[base + tid], v1 = in[base + tid + 256];
  if (has_res) { v0 += res[base + tid]; v1 += res[base + tid + 256]; }
  float s = v0 + v1;
  #pragma unroll
  for (int o = 16; o > 0; o >>= 1) s += __shfl_xor(s, o, 32);
  if ((tid & 31) == 0) red[tid >> 5] = s;
  __syncthreads();
  if (tid == 0) { float a = 0; for (int i = 0; i < 8; ++i) a += red[i]; red[8] = a * (1.0f / DMODEL); }
  __syncthreads();
  float mean = red[8];
  float d0 = v0 - mean, d1 = v1 - mean;
  float q = d0 * d0 + d1 * d1;
  #pragma unroll
  for (int o = 16; o > 0; o >>= 1) q += __shfl_xor(q, o, 32);
  __syncthreads();
  if ((tid & 31) == 0) red[tid >> 5] = q;
  __syncthreads();
  if (tid == 0) { float a = 0; for (int i = 0; i < 8; ++i) a += red[i]; red[9] = rsqrtf(a * (1.0f / DMODEL) + eps); }
  __syncthreads();
  float rstd = red[9];
  float o0 = d0 * rstd * g[tid] + b[tid];
  float o1 = d1 * rstd * g[tid + 256] + b[tid + 256];
  out[base + tid] = o0;
  out[base + tid + 256] = o1;
  if (obf) { obf[base + tid] = f2bf(o0); obf[base + tid + 256] = f2bf(o1); }
}

// ---------------- small elementwise kernels ----------------
__global__ void cvt_bf_k(const float* __restrict__ s, bf16_t* __restrict__ d, int n) {
  int i = blockIdx.x * blockDim.x + threadIdx.x;
  if (i < n) d[i] = f2bf(s[i]);
}
__global__ void neg_exp_k(const float* __restrict__ a, float* __restrict__ o, int n) {
  int i = blockIdx.x * blockDim.x + threadIdx.x;
  if (i < n) o[i] = -__expf(a[i]);
}
__global__ void softplus_k(float* __restrict__ x, int n) {
  int i = blockIdx.x * blockDim.x + threadIdx.x;
  if (i < n) { float v = x[i]; x[i] = (v > 20.0f) ? v : log1pf(__expf(v)); }
}
__global__ void split_dt_k(const float* __restrict__ xdbl, bf16_t* __restrict__ dtbf) {
  int i = blockIdx.x * blockDim.x + threadIdx.x;
  if (i >= NT * 32) return;
  int t = i >> 5, j = i & 31;
  dtbf[i] = f2bf(xdbl[(size_t)t * 64 + j]);
}
// xg = h1 * silu(h2); writes bf16 forward + time-reversed copies
__global__ void gate_silu_rev_k(const float* __restrict__ h1, const float* __restrict__ h2,
                                bf16_t* __restrict__ xgbf, bf16_t* __restrict__ xgrbf) {
  size_t id = (size_t)blockIdx.x * blockDim.x + threadIdx.x;
  if (id >= (size_t)NT * DMODEL) return;
  int j = (int)(id & (DMODEL - 1));
  int t = (int)(id >> 9);
  float v = h1[id] * siluf_(h2[id]);
  bf16_t bv = f2bf(v);
  xgbf[id] = bv;
  int bb = t >> 11, l = t & (SEQL - 1);
  xgrbf[((size_t)(bb * SEQL + (SEQL - 1 - l))) * DMODEL + j] = bv;
}
// out2 = ha*silu(hbp) + ha + out1
__global__ void gate2_k(const float* __restrict__ ha, const float* __restrict__ hbp,
                        const float* __restrict__ out1, float* __restrict__ out2) {
  size_t id = (size_t)blockIdx.x * blockDim.x + threadIdx.x;
  if (id >= (size_t)NT * DMODEL) return;
  float a = ha[id];
  out2[id] = a * siluf_(hbp[id]) + a + out1[id];
}
__global__ void gelu_bf_k(const float* __restrict__ x, bf16_t* __restrict__ o, int n) {
  int i = blockIdx.x * blockDim.x + threadIdx.x;
  if (i < n) { float v = x[i]; o[i] = f2bf(0.5f * v * (1.0f + erff(v * 0.70710678118654752f))); }
}
// depthwise causal conv (k=4) + SiLU over x-half of xz; writes f32 + bf16
__global__ void conv_silu_k(const float* __restrict__ xz, const float* __restrict__ cw,
                            const float* __restrict__ cb,
                            float* __restrict__ xc, bf16_t* __restrict__ xcbf) {
  size_t id = (size_t)blockIdx.x * blockDim.x + threadIdx.x;
  if (id >= (size_t)NT * DINNER) return;
  int ch = (int)(id & (DINNER - 1));
  int t  = (int)(id >> 10);
  int l  = t & (SEQL - 1);
  int bb = t >> 11;
  float acc = cb[ch];
  #pragma unroll
  for (int k = 0; k < 4; ++k) {
    int lk = l - 3 + k;
    if (lk >= 0) acc += cw[ch * 4 + k] * xz[((size_t)(bb * SEQL + lk)) * 2048 + ch];
  }
  float v = siluf_(acc);
  xc[id] = v;
  xcbf[id] = f2bf(v);
}

// =====================================================================
// Chunked selective scan: h_t = exp(d_t*A)*h_{t-1} + d_t*u_t*B_t  (16 states)
// pass1: per-chunk (P = prod a, R = scan from 0); layout [chunk][b][ch][n]
// pass2: serial over 32 chunks -> chunk-initial states H
// pass3: replay with true h0, y_t = sum_n h*C; fused (y + u*D)*silu(z) -> bf16
// =====================================================================
__global__ void scan_pass1_k(const float* __restrict__ delta, const float* __restrict__ xc,
                             const float* __restrict__ xdbl, const float* __restrict__ Aexp,
                             float* __restrict__ P, float* __restrict__ R) {
  int id = blockIdx.x * blockDim.x + threadIdx.x;
  if (id >= NCHUNK * BATCHN * DINNER) return;
  int ch = id & (DINNER - 1);
  int bb = (id >> 10) & 3;
  int ck = id >> 12;
  float p[DSTATE], r[DSTATE], Av[DSTATE];
  #pragma unroll
  for (int n = 0; n < DSTATE; ++n) { Av[n] = Aexp[ch * DSTATE + n]; p[n] = 1.0f; r[n] = 0.0f; }
  for (int i = 0; i < CLEN; ++i) {
    int t = bb * SEQL + ck * CLEN + i;
    float d  = delta[(size_t)t * DINNER + ch];
    float du = d * xc[(size_t)t * DINNER + ch];
    const float* Bm = xdbl + (size_t)t * 64 + 32;
    #pragma unroll
    for (int n = 0; n < DSTATE; ++n) {
      float dA = __expf(d * Av[n]);
      p[n] *= dA;
      r[n] = dA * r[n] + du * Bm[n];
    }
  }
  size_t o = (size_t)id * DSTATE;
  #pragma unroll
  for (int n = 0; n < DSTATE; ++n) { P[o + n] = p[n]; R[o + n] = r[n]; }
}

__global__ void scan_pass2_k(const float* __restrict__ P, const float* __restrict__ R,
                             float* __restrict__ H) {
  int id = blockIdx.x * blockDim.x + threadIdx.x;      // (b*1024+ch)*16+n
  if (id >= BATCHN * DINNER * DSTATE) return;
  float h = 0.0f;
  for (int c = 0; c < NCHUNK; ++c) {
    size_t idx = (size_t)c * (BATCHN * DINNER * DSTATE) + id;
    H[idx] = h;
    h = P[idx] * h + R[idx];
  }
}

__global__ void scan_pass3_k(const float* __restrict__ delta, const float* __restrict__ xc,
                             const float* __restrict__ xdbl, const float* __restrict__ Aexp,
                             const float* __restrict__ H, const float* __restrict__ xz,
                             const float* __restrict__ Dp, bf16_t* __restrict__ ybf) {
  int id = blockIdx.x * blockDim.x + threadIdx.x;
  if (id >= NCHUNK * BATCHN * DINNER) return;
  int ch = id & (DINNER - 1);
  int bb = (id >> 10) & 3;
  int ck = id >> 12;
  float h[DSTATE], Av[DSTATE];
  size_t o = (size_t)id * DSTATE;
  #pragma unroll
  for (int n = 0; n < DSTATE; ++n) { h[n] = H[o + n]; Av[n] = Aexp[ch * DSTATE + n]; }
  float Dv = Dp[ch];
  for (int i = 0; i < CLEN; ++i) {
    int t = bb * SEQL + ck * CLEN + i;
    float d  = delta[(size_t)t * DINNER + ch];
    float u  = xc[(size_t)t * DINNER + ch];
    float du = d * u;
    const float* Bm = xdbl + (size_t)t * 64 + 32;
    const float* Cm = xdbl + (size_t)t * 64 + 48;
    float y = 0.0f;
    #pragma unroll
    for (int n = 0; n < DSTATE; ++n) {
      float dA = __expf(d * Av[n]);
      h[n] = dA * h[n] + du * Bm[n];
      y += h[n] * Cm[n];
    }
    float z = xz[(size_t)t * 2048 + 1024 + ch];
    ybf[(size_t)t * DINNER + ch] = f2bf((y + u * Dv) * siluf_(z));
  }
}

// =====================================================================
// host-side orchestration
// =====================================================================
extern "C" void kernel_launch(void* const* d_in, const int* in_sizes, int n_in,
                              void* d_out, int out_size, void* d_ws, size_t ws_size,
                              hipStream_t stream) {
  (void)in_sizes; (void)n_in; (void)out_size; (void)ws_size;

  const float* x     = (const float*)d_in[0];
  const float* fre   = (const float*)d_in[1];
  const float* fim   = (const float*)d_in[2];
  const float* cut   = (const float*)d_in[3];
  const float* ln0g  = (const float*)d_in[4];
  const float* ln0b  = (const float*)d_in[5];
  const float* l1w   = (const float*)d_in[6];
  const float* l1b   = (const float*)d_in[7];
  const float* l2w   = (const float*)d_in[8];
  const float* l2b   = (const float*)d_in[9];
  const float* inw   = (const float*)d_in[10];
  const float* convw = (const float*)d_in[11];
  const float* convb = (const float*)d_in[12];
  const float* xpw   = (const float*)d_in[13];
  const float* dtw   = (const float*)d_in[14];
  const float* dtb   = (const float*)d_in[15];
  const float* Alog  = (const float*)d_in[16];
  const float* Dp    = (const float*)d_in[17];
  const float* outw  = (const float*)d_in[18];
  const float* ln1g  = (const float*)d_in[19];
  const float* ln1b  = (const float*)d_in[20];
  const float* d1w   = (const float*)d_in[21];
  const float* d1b   = (const float*)d_in[22];
  const float* d2w   = (const float*)d_in[23];
  const float* d2b   = (const float*)d_in[24];
  const float* ln2g  = (const float*)d_in[25];
  const float* ln2b  = (const float*)d_in[26];
  const float* f1w   = (const float*)d_in[27];
  const float* f1b   = (const float*)d_in[28];
  const float* f2w   = (const float*)d_in[29];
  const float* f2b   = (const float*)d_in[30];
  const float* lfg   = (const float*)d_in[31];
  const float* lfb   = (const float*)d_in[32];
  float* out = (float*)d_out;

  // ---- workspace bump allocator (256B aligned) ----
  char* wsb = (char*)d_ws;
  size_t off = 0;
  auto alloc = [&](size_t bytes) -> char* {
    off = (off + 255) & ~(size_t)255;
    char* p = wsb + off;
    off += bytes;
    return p;
  };
  // bf16 weights
  bf16_t* wM   = (bf16_t*)alloc((size_t)512 * 512 * 2);
  bf16_t* wl1  = (bf16_t*)alloc((size_t)512 * 512 * 2);
  bf16_t* wl2  = (bf16_t*)alloc((size_t)512 * 512 * 2);
  bf16_t* win  = (bf16_t*)alloc((size_t)2048 * 512 * 2);
  bf16_t* wxp  = (bf16_t*)alloc((size_t)64 * 1024 * 2);
  bf16_t* wdt  = (bf16_t*)alloc((size_t)1024 * 32 * 2);
  bf16_t* wout = (bf16_t*)alloc((size_t)512 * 1024 * 2);
  bf16_t* wd1  = (bf16_t*)alloc((size_t)512 * 512 * 2);
  bf16_t* wd2  = (bf16_t*)alloc((size_t)512 * 512 * 2);
  bf16_t* wf1  = (bf16_t*)alloc((size_t)2048 * 512 * 2);
  bf16_t* wf2  = (bf16_t*)alloc((size_t)512 * 2048 * 2);
  float*  aexp = (float*)alloc((size_t)DINNER * DSTATE * 4);
  // bf16 activations
  bf16_t* xbf   = (bf16_t*)alloc((size_t)NT * 512 * 2);
  bf16_t* xnbf  = (bf16_t*)alloc((size_t)NT * 512 * 2);
  bf16_t* obf   = (bf16_t*)alloc((size_t)NT * 512 * 2);
  bf16_t* hsbf  = (bf16_t*)alloc((size_t)NT * 512 * 2);
  bf16_t* xgbf  = (bf16_t*)alloc((size_t)NT * 512 * 2);
  bf16_t* xgrbf = (bf16_t*)alloc((size_t)NT * 512 * 2);
  // f32 activation pool
  float* fA = (float*)alloc((size_t)NT * 512 * 4);
  float* fB = (float*)alloc((size_t)NT * 512 * 4);
  float* fC = (float*)alloc((size_t)NT * 512 * 4);
  float* fD = (float*)alloc((size_t)NT * 512 * 4);
  float* fE = (float*)alloc((size_t)NT * 512 * 4);
  float* xz = (float*)alloc((size_t)NT * 2048 * 4);   // also FFN hidden f32
  float* xc = (float*)alloc((size_t)NT * DINNER * 4); // also FFN hidden bf16 region
  bf16_t* xcbf = (bf16_t*)alloc((size_t)NT * DINNER * 2);
  float* xdbl  = (float*)alloc((size_t)NT * 64 * 4);
  bf16_t* dtbf = (bf16_t*)alloc((size_t)NT * 32 * 2);
  float* delta = (float*)alloc((size_t)NT * DINNER * 4);
  bf16_t* ybf  = (bf16_t*)alloc((size_t)NT * DINNER * 2);
  const size_t scsz = (size_t)NCHUNK * BATCHN * DINNER * DSTATE;
  float* Pc = (float*)alloc(scsz * 4);
  float* Rc = (float*)alloc(scsz * 4);
  float* Hc = (float*)alloc(scsz * 4);

  auto GEMM = [&](const bf16_t* A, const bf16_t* B, const float* bias, float* C,
                  int M, int N, int K, int flags) {
    dim3 g(N / BN, M / BM);
    gemm_bf16_wmma<<<g, 128, 0, stream>>>(A, B, bias, C, M, N, K, flags);
  };
  auto CVT = [&](const float* s, bf16_t* d, int n) {
    cvt_bf_k<<<(n + 255) / 256, 256, 0, stream>>>(s, d, n);
  };

  // ---- weight prep ----
  build_circ_k<<<1, 512, 0, stream>>>(fre, fim, cut, wM);
  CVT(l1w, wl1, 512 * 512);
  CVT(l2w, wl2, 512 * 512);
  CVT(inw, win, 2048 * 512);
  CVT(xpw, wxp, 64 * 1024);
  CVT(dtw, wdt, 1024 * 32);
  CVT(outw, wout, 512 * 1024);
  CVT(d1w, wd1, 512 * 512);
  CVT(d2w, wd2, 512 * 512);
  CVT(f1w, wf1, 2048 * 512);
  CVT(f2w, wf2, 512 * 2048);
  CVT(x, xbf, NT * 512);
  neg_exp_k<<<(DINNER * DSTATE + 255) / 256, 256, 0, stream>>>(Alog, aexp, DINNER * DSTATE);

  // ---- spectral filter (as circulant GEMM) + residual + LN0 ----
  GEMM(xbf, wM, nullptr, fA, NT, 512, 512, 0);
  layernorm_k<<<NT, 256, 0, stream>>>(fA, x, ln0g, ln0b, fB, xnbf, 1e-5f, 1);

  // ---- gating branch: xg = (xn@l1+b) * silu(xn@l2+b) ----
  GEMM(xnbf, wl1, l1b, fC, NT, 512, 512, 1);
  GEMM(xnbf, wl2, l2b, fD, NT, 512, 512, 1);
  {
    size_t n = (size_t)NT * 512;
    gate_silu_rev_k<<<(unsigned)((n + 255) / 256), 256, 0, stream>>>(fC, fD, xgbf, xgrbf);
  }

  // ---- mamba block (shared weights), accumulate into fE ----
  auto run_mamba = [&](const bf16_t* inbf, int accum) {
    GEMM(inbf, win, nullptr, xz, NT, 2048, 512, 0);                  // in-proj
    {
      size_t n = (size_t)NT * DINNER;
      conv_silu_k<<<(unsigned)((n + 255) / 256), 256, 0, stream>>>(xz, convw, convb, xc, xcbf);
    }
    GEMM(xcbf, wxp, nullptr, xdbl, NT, 64, 1024, 0);                 // x_proj
    split_dt_k<<<(NT * 32 + 255) / 256, 256, 0, stream>>>(xdbl, dtbf);
    GEMM(dtbf, wdt, dtb, delta, NT, 1024, 32, 1);                    // dt-proj + bias
    softplus_k<<<(NT * DINNER + 255) / 256, 256, 0, stream>>>(delta, NT * DINNER);
    int sc = NCHUNK * BATCHN * DINNER;
    scan_pass1_k<<<(sc + 255) / 256, 256, 0, stream>>>(delta, xc, xdbl, aexp, Pc, Rc);
    scan_pass2_k<<<(BATCHN * DINNER * DSTATE + 255) / 256, 256, 0, stream>>>(Pc, Rc, Hc);
    scan_pass3_k<<<(sc + 255) / 256, 256, 0, stream>>>(delta, xc, xdbl, aexp, Hc, xz, Dp, ybf);
    GEMM(ybf, wout, nullptr, fE, NT, 512, 1024, accum ? 2 : 0);      // out-proj (+=)
  };
  run_mamba(xgbf, 0);
  run_mamba(xgrbf, 1);
  run_mamba(xbf, 1);

  // ---- LN1, dual gate, LN2 ----
  layernorm_k<<<NT, 256, 0, stream>>>(fE, nullptr, ln1g, ln1b, fA, obf, 1e-12f, 0);
  GEMM(obf, wd1, d1b, fB, NT, 512, 512, 1);   // hb_pre
  GEMM(obf, wd2, d2b, fC, NT, 512, 512, 1);   // ha
  {
    size_t n = (size_t)NT * 512;
    gate2_k<<<(unsigned)((n + 255) / 256), 256, 0, stream>>>(fC, fB, fA, fD);
  }
  layernorm_k<<<NT, 256, 0, stream>>>(fD, nullptr, ln2g, ln2b, fB, hsbf, 1e-12f, 0);

  // ---- FFN ----
  GEMM(hsbf, wf1, f1b, xz, NT, 2048, 512, 1);
  bf16_t* fhbf = (bf16_t*)xc;  // reuse xc region (N*1024 f32 == N*2048 bf16)
  gelu_bf_k<<<(NT * 2048 + 255) / 256, 256, 0, stream>>>(xz, fhbf, NT * 2048);
  GEMM(fhbf, wf2, f2b, fC, NT, 512, 2048, 1);

  // ---- final LN(h + hs) -> d_out ----
  layernorm_k<<<NT, 256, 0, stream>>>(fC, fB, lfg, lfb, out, nullptr, 1e-12f, 1);
}
